// GroupEnhance_35854386987428
// MI455X (gfx1250) — matrix operand order repaired and checked by reading, not compiled, not run on hardware
//
#include <hip/hip_runtime.h>
#include <hip/hip_bf16.h>

#define DIM 512
#define LDA 520  // LDS row stride in halfs: 1040 B = 260 dwords -> rows land 4 banks apart, conflict-free b128 reads

typedef __attribute__((ext_vector_type(16))) _Float16 v16h;
typedef __attribute__((ext_vector_type(8)))  _Float16 v8h;
typedef __attribute__((ext_vector_type(8)))  float    v8f;
typedef __attribute__((ext_vector_type(4)))  float    v4f;
typedef __attribute__((ext_vector_type(4)))  _Float16 v4h;

// ---------------- kernel 1: zero agg + deg ----------------
__global__ __launch_bounds__(256) void ge_zero_kernel(float* p, long long total) {
    long long i = (long long)blockIdx.x * blockDim.x + threadIdx.x;
    long long stride = (long long)gridDim.x * blockDim.x;
    for (; i < total; i += stride) p[i] = 0.0f;
}

// ---------------- kernel 2: warp-per-edge scatter-add ----------------
__global__ __launch_bounds__(256) void ge_scatter_kernel(const float* __restrict__ x,
                                                         const int* __restrict__ src,
                                                         const int* __restrict__ dst,
                                                         float* __restrict__ agg,
                                                         float* __restrict__ deg,
                                                         int nedges) {
    int warp = (blockIdx.x * blockDim.x + threadIdx.x) >> 5;
    int lane = threadIdx.x & 31;
    if (warp >= nedges) return;
    int s = src[warp];
    int d = dst[warp];
    const float* xr = x + (size_t)s * DIM;
    float* ar = agg + (size_t)d * DIM;
    #pragma unroll
    for (int k = lane; k < DIM; k += 32) atomicAdd(&ar[k], xr[k]);
    if (lane == 0) atomicAdd(&deg[d], 1.0f);
}

// ---------------- kernel 3a: agg/deg -> f16 ----------------
__global__ __launch_bounds__(256) void ge_norm_cvt_kernel(const float* __restrict__ agg,
                                                          const float* __restrict__ deg,
                                                          _Float16* __restrict__ agg_h,
                                                          long long total4) {
    long long i = (long long)blockIdx.x * blockDim.x + threadIdx.x;
    long long stride = (long long)gridDim.x * blockDim.x;
    for (; i < total4; i += stride) {
        long long row = i >> 7;  // 128 float4 groups per 512-wide row
        float dv = deg[row];
        float inv = 1.0f / fmaxf(dv, 1.0f);
        v4f v = ((const v4f*)agg)[i];
        v4h h;
        h.x = (_Float16)(v.x * inv);
        h.y = (_Float16)(v.y * inv);
        h.z = (_Float16)(v.z * inv);
        h.w = (_Float16)(v.w * inv);
        ((v4h*)agg_h)[i] = h;
    }
}

// ---------------- kernel 3b: W -> f16 ----------------
__global__ __launch_bounds__(256) void ge_cvt_kernel(const float* __restrict__ W,
                                                     _Float16* __restrict__ W_h,
                                                     long long total4) {
    long long i = (long long)blockIdx.x * blockDim.x + threadIdx.x;
    long long stride = (long long)gridDim.x * blockDim.x;
    for (; i < total4; i += stride) {
        v4f v = ((const v4f*)W)[i];
        v4h h;
        h.x = (_Float16)v.x; h.y = (_Float16)v.y;
        h.z = (_Float16)v.z; h.w = (_Float16)v.w;
        ((v4h*)W_h)[i] = h;
    }
}

// ---------------- kernel 4: WMMA GEMM + residual epilogue ----------------
// Block: 256 threads (8 waves). One block owns a 16-row M-tile, full N=512.
// A-slab (16 x 512 f16) staged in LDS once; wave w computes N in [64w, 64w+64).
__global__ __launch_bounds__(256) void ge_gemm_kernel(const _Float16* __restrict__ agg_h,
                                                      const _Float16* __restrict__ W_h,
                                                      const float* __restrict__ x,
                                                      const float* __restrict__ bias,
                                                      const float* __restrict__ alpha_p,
                                                      float* __restrict__ out,
                                                      int nrows) {
    __shared__ _Float16 Atile[16 * LDA];

    const int tid   = threadIdx.x;
    const int mbase = blockIdx.x * 16;

    // cooperative load of the 16x512 A slab (16B chunks; 64 chunks per row)
    for (int c = tid; c < 16 * 64; c += 256) {
        int row = c >> 6;
        int col = c & 63;
        uint4 v = {0u, 0u, 0u, 0u};
        if (mbase + row < nrows)
            v = ((const uint4*)(agg_h + (size_t)(mbase + row) * DIM))[col];
        *(uint4*)(Atile + row * LDA + col * 8) = v;
    }
    __syncthreads();

    const int wid     = tid >> 5;        // wave id: N slice [64*wid, 64*wid+64)
    const int lane    = tid & 31;
    const int half_id = lane >> 4;       // 0: lanes 0-15, 1: lanes 16-31
    const int m       = lane & 15;

    const int nbase0 = wid * 64;

    v8f acc0 = {0.f,0.f,0.f,0.f,0.f,0.f,0.f,0.f};
    v8f acc1 = acc0, acc2 = acc0, acc3 = acc0;

    const _Float16* arow = Atile + m * LDA;
    const _Float16* brow0 = W_h + (size_t)(nbase0 +  0 + m) * DIM;
    const _Float16* brow1 = W_h + (size_t)(nbase0 + 16 + m) * DIM;
    const _Float16* brow2 = W_h + (size_t)(nbase0 + 32 + m) * DIM;
    const _Float16* brow3 = W_h + (size_t)(nbase0 + 48 + m) * DIM;

    #pragma unroll 4
    for (int kk = 0; kk < DIM; kk += 32) {
        // A fragment: lo-lanes K0..15, hi-lanes K16..31 (contiguous, matches B)
        v8h a0 = *(const v8h*)(arow + kk + half_id * 16);
        v8h a1 = *(const v8h*)(arow + kk + half_id * 16 + 8);
        v16h a = __builtin_shufflevector(a0, a1, 0,1,2,3,4,5,6,7,8,9,10,11,12,13,14,15);

        // B fragment: lane n holds 16 contiguous K values (lo-lanes K0..15, hi-lanes K16..31)
        v16h b0 = *(const v16h*)(brow0 + kk + half_id * 16);
        v16h b1 = *(const v16h*)(brow1 + kk + half_id * 16);
        v16h b2 = *(const v16h*)(brow2 + kk + half_id * 16);
        v16h b3 = *(const v16h*)(brow3 + kk + half_id * 16);

        acc0 = __builtin_amdgcn_wmma_f32_16x16x32_f16(false, a, false, b0, (short)0, acc0, false, false);
        acc1 = __builtin_amdgcn_wmma_f32_16x16x32_f16(false, a, false, b1, (short)0, acc1, false, false);
        acc2 = __builtin_amdgcn_wmma_f32_16x16x32_f16(false, a, false, b2, (short)0, acc2, false, false);
        acc3 = __builtin_amdgcn_wmma_f32_16x16x32_f16(false, a, false, b3, (short)0, acc3, false, false);
    }

    const float alpha = alpha_p[0];

    // C/D layout: VGPR r -> row (half_id*8 + r), col = lane%16
    v8f accs[4] = {acc0, acc1, acc2, acc3};
    #pragma unroll
    for (int t = 0; t < 4; ++t) {
        int col = nbase0 + t * 16 + m;
        float bv = bias[col];
        #pragma unroll
        for (int r = 0; r < 8; ++r) {
            int row = mbase + half_id * 8 + r;
            if (row < nrows) {
                size_t idx = (size_t)row * DIM + col;
                out[idx] = x[idx] + alpha * (accs[t][r] + bv);
            }
        }
    }
}

// ---------------- host launcher ----------------
extern "C" void kernel_launch(void* const* d_in, const int* in_sizes, int n_in,
                              void* d_out, int out_size, void* d_ws, size_t ws_size,
                              hipStream_t stream) {
    const float* x       = (const float*)d_in[0];
    const int*   edges   = (const int*)d_in[1];
    const float* alpha_p = (const float*)d_in[3];
    const float* W       = (const float*)d_in[4];
    const float* bias    = (const float*)d_in[5];

    const int N = in_sizes[0] / DIM;   // 50000
    const int E = in_sizes[1] / 2;     // 150000
    const int* src = edges;
    const int* dst = edges + E;

    // workspace layout: agg f32 | deg f32 | agg f16 | W f16
    char* ws = (char*)d_ws;
    float* agg = (float*)ws;
    size_t off = (size_t)N * DIM * sizeof(float);
    float* deg = (float*)(ws + off);
    off += ((size_t)N * sizeof(float) + 255) & ~(size_t)255;
    _Float16* agg_h = (_Float16*)(ws + off);
    off += (size_t)N * DIM * sizeof(_Float16);
    _Float16* W_h = (_Float16*)(ws + off);

    // 1) zero agg + deg (deg is adjacent-but-padded; zero both regions)
    long long ztot = (long long)N * DIM;
    ge_zero_kernel<<<(int)((ztot + 255) / 256), 256, 0, stream>>>(agg, ztot);
    ge_zero_kernel<<<(N + 255) / 256, 256, 0, stream>>>(deg, N);

    // 2) scatter-add (warp per edge)
    {
        long long warps = E;
        int blocks = (int)((warps * 32 + 255) / 256);
        ge_scatter_kernel<<<blocks, 256, 0, stream>>>(x, src, dst, agg, deg, E);
    }

    // 3) normalize+convert agg, convert W
    {
        long long t4 = (long long)N * DIM / 4;
        ge_norm_cvt_kernel<<<(int)((t4 + 255) / 256), 256, 0, stream>>>(agg, deg, agg_h, t4);
        long long w4 = (long long)DIM * DIM / 4;
        ge_cvt_kernel<<<(int)((w4 + 255) / 256), 256, 0, stream>>>(W, W_h, w4);
    }

    // 4) WMMA GEMM + epilogue
    {
        int mblocks = (N + 15) / 16;   // 3125
        ge_gemm_kernel<<<mblocks, 256, 0, stream>>>(agg_h, W_h, x, bias, alpha_p,
                                                    (float*)d_out, N);
    }
}
